// label_swapper_dynamic_85538568667217
// MI455X (gfx1250) — matrix operand order repaired and checked by reading, not compiled, not run on hardware
//
#include <hip/hip_runtime.h>
#include <hip/hip_bf16.h>

typedef __attribute__((ext_vector_type(16))) _Float16 v16h;
typedef __attribute__((ext_vector_type(8)))  _Float16 v8h;
typedef __attribute__((ext_vector_type(8)))  float    v8f;

#define N_TAB   200000
#define BATCH   2048
#define NCLS    100
#define LAT     20
#define NTILES  (N_TAB / 16)      // 12500
#define NSPLIT  10
#define NT_PER  (NTILES / NSPLIT) // 1250
#define WAVES   8
#define PKW     32                // packed row width in halfs (64 B)
#define FLT_BIG 3.402823e38f

// ---------------- Phase 0a (packed path): f16-pack table rows + fold in ||t||^2 ----------------
// Packed row r (32 halfs): [0..19] = t_k, [20] = ||t||^2, [21..31] = 0.
__global__ void pack_kernel(const float* __restrict__ table, _Float16* __restrict__ pk) {
    int r = blockIdx.x * blockDim.x + threadIdx.x;
    if (r >= N_TAB) return;
    const float* p = table + r * LAT;
    float v[LAT], s = 0.f;
#pragma unroll
    for (int k = 0; k < LAT; ++k) { v[k] = p[k]; s = fmaf(v[k], v[k], s); }
    v8h o0, o1, o2, o3;
#pragma unroll
    for (int i = 0; i < 8; ++i) { o0[i] = (_Float16)v[i]; o1[i] = (_Float16)v[8 + i]; }
    o2[0] = (_Float16)v[16]; o2[1] = (_Float16)v[17];
    o2[2] = (_Float16)v[18]; o2[3] = (_Float16)v[19];
    o2[4] = (_Float16)s;                       // K=20 slot: row norm
    o2[5] = o2[6] = o2[7] = (_Float16)0.f;
#pragma unroll
    for (int i = 0; i < 8; ++i) o3[i] = (_Float16)0.f;
    v8h* dst = (v8h*)(pk + (size_t)r * PKW);   // 64B rows, 16B-aligned chunks
    dst[0] = o0; dst[1] = o1; dst[2] = o2; dst[3] = o3;
}

// ---------------- Phase 0b (fallback path): table row squared norms ----------------
__global__ void tn_kernel(const float* __restrict__ table, float* __restrict__ tn) {
    int i = blockIdx.x * blockDim.x + threadIdx.x;
    if (i < N_TAB) {
        const float* p = table + i * LAT;
        float s = 0.f;
#pragma unroll
        for (int k = 0; k < LAT; ++k) s = fmaf(p[k], p[k], s);
        tn[i] = s;
    }
}

// Shared epilogue: per-(query,chunk) argmin reduction across waves/lanes.
__device__ __forceinline__ void reduce_and_emit(
    float (&sMin)[WAVES][8][32], int (&sIdx)[WAVES][8][32],
    int tid, int mtile, int chunk, float* pmin, int* pidx)
{
    if (tid < 16) {
        const int m = tid, v = m & 7, lbase = (m >> 3) * 16;
        float best = FLT_BIG; int bidx = 0x7fffffff;
        for (int w = 0; w < WAVES; ++w)
            for (int l = 0; l < 16; ++l) {
                float val = sMin[w][v][lbase + l];
                int   id  = sIdx[w][v][lbase + l];
                if (val < best || (val == best && id < bidx)) { best = val; bidx = id; }
            }
        const int qg = mtile * 16 + m;
        pmin[qg * NSPLIT + chunk] = best;
        pidx[qg * NSPLIT + chunk] = bidx;
    }
}

// ---------------- Phase 1 (packed): 2 tiles in flight, score straight out of WMMA ----------------
__global__ __launch_bounds__(256) void knn_phase1_packed(
    const float*    __restrict__ mius,  // [BATCH][LAT]
    const _Float16* __restrict__ pk,    // [N_TAB][PKW] packed f16 rows (incl. norm at K=20)
    float* __restrict__ pmin, int* __restrict__ pidx)
{
    __shared__ _Float16 ldsA[16][32];
    __shared__ float sMin[WAVES][8][32];
    __shared__ int   sIdx[WAVES][8][32];

    const int tid   = threadIdx.x;
    const int lane  = tid & 31;
    const int wave  = tid >> 5;
    const int mtile = blockIdx.x;
    const int chunk = blockIdx.y;

    // Stage A = queries scaled by -2 (exact in f16), with 1.0 at K=20 to pick up ||t||^2.
    const float* q = mius + mtile * 16 * LAT;
    for (int i = tid; i < 16 * 32; i += 256) {
        int m = i >> 5, k = i & 31;
        float v = (k < LAT) ? -2.f * q[m * LAT + k] : (k == LAT ? 1.f : 0.f);
        ldsA[m][k] = (_Float16)v;
    }
    __syncthreads();

    // A fragment (contiguous K order): lo lanes K0..15, hi lanes K16..31.
    const int  mrow = lane & 15;
    const bool lo   = lane < 16;
    const int  kb0  = lo ? 0 : 16;
    const int  kb1  = lo ? 8 : 24;
    v16h Av;
#pragma unroll
    for (int i = 0; i < 8; ++i) Av[i]     = ldsA[mrow][kb0 + i];
#pragma unroll
    for (int i = 0; i < 8; ++i) Av[8 + i] = ldsA[mrow][kb1 + i];

    // Per-lane running argmin: slot v <-> (M = v + 8*(lane>=16), table row = tile*16 + (lane&15)).
    float mn[8];
    int   ix[8];
#pragma unroll
    for (int v = 0; v < 8; ++v) { mn[v] = FLT_BIG; ix[v] = 0; }

    const int khalf = lo ? 0 : 16;               // blocked B: lo lanes K0..15, hi lanes K16..31
    const int tEnd  = (chunk + 1) * NT_PER;

    // Manual 2-way unroll: two independent load->WMMA->argmin chains per pass so the
    // scheduler overlaps B loads with the other tile's WMMA and fills the WMMA->VALU
    // hazard slots. If the second index clamps onto a duplicate tile, the strict-<
    // update makes reprocessing a no-op (no remainder branch needed).
    for (int t = chunk * NT_PER + wave; t < tEnd; t += 2 * WAVES) {
        int t2 = t + WAVES;
        if (t2 >= tEnd) t2 = t;
        const int row0 = t  * 16 + mrow;
        const int row1 = t2 * 16 + mrow;
        const _Float16* bp0 = pk + (size_t)row0 * PKW + khalf;
        const _Float16* bp1 = pk + (size_t)row1 * PKW + khalf;
        __builtin_prefetch(bp0 + 2 * WAVES * 16 * PKW, 0, 3);
        __builtin_prefetch(bp1 + 2 * WAVES * 16 * PKW, 0, 3);

        const v8h a0 = ((const v8h*)bp0)[0];     // 16B-aligned b128 loads
        const v8h a1 = ((const v8h*)bp0)[1];
        const v8h c0 = ((const v8h*)bp1)[0];
        const v8h c1 = ((const v8h*)bp1)[1];
        v16h B0, B1;
#pragma unroll
        for (int i = 0; i < 8; ++i) {
            B0[i] = a0[i]; B0[8 + i] = a1[i];
            B1[i] = c0[i]; B1[8 + i] = c1[i];
        }

        v8f acc0 = {}, acc1 = {};
        acc0 = __builtin_amdgcn_wmma_f32_16x16x32_f16(false, Av, false, B0, (short)0, acc0, false, false);
        acc1 = __builtin_amdgcn_wmma_f32_16x16x32_f16(false, Av, false, B1, (short)0, acc1, false, false);
        // acc[v] == ||t||^2 - 2 q.t  (score), norm folded in via K=20 lane.
#pragma unroll
        for (int v = 0; v < 8; ++v)
            if (acc0[v] < mn[v]) { mn[v] = acc0[v]; ix[v] = row0; }
#pragma unroll
        for (int v = 0; v < 8; ++v)
            if (acc1[v] < mn[v]) { mn[v] = acc1[v]; ix[v] = row1; }
    }

#pragma unroll
    for (int v = 0; v < 8; ++v) { sMin[wave][v][lane] = mn[v]; sIdx[wave][v][lane] = ix[v]; }
    __syncthreads();
    reduce_and_emit(sMin, sIdx, tid, mtile, chunk, pmin, pidx);
}

// ---------------- Phase 1 (fallback): on-the-fly f32->f16 conversion, tn from global ----------------
__global__ __launch_bounds__(256) void knn_phase1_fallback(
    const float* __restrict__ mius, const float* __restrict__ table,
    const float* __restrict__ tn, float* __restrict__ pmin, int* __restrict__ pidx)
{
    __shared__ _Float16 ldsA[16][32];
    __shared__ float sMin[WAVES][8][32];
    __shared__ int   sIdx[WAVES][8][32];

    const int tid = threadIdx.x, lane = tid & 31, wave = tid >> 5;
    const int mtile = blockIdx.x, chunk = blockIdx.y;

    const float* q = mius + mtile * 16 * LAT;
    for (int i = tid; i < 16 * 32; i += 256) {
        int m = i >> 5, k = i & 31;
        ldsA[m][k] = (k < LAT) ? (_Float16)q[m * LAT + k] : (_Float16)0.f;
    }
    __syncthreads();

    const int mrow = lane & 15;
    const bool lo  = lane < 16;
    const int kb0 = lo ? 0 : 16, kb1 = lo ? 8 : 24;
    v16h Av;
#pragma unroll
    for (int i = 0; i < 8; ++i) Av[i]     = ldsA[mrow][kb0 + i];
#pragma unroll
    for (int i = 0; i < 8; ++i) Av[8 + i] = ldsA[mrow][kb1 + i];

    float mn[8]; int ix[8];
#pragma unroll
    for (int v = 0; v < 8; ++v) { mn[v] = FLT_BIG; ix[v] = 0; }

    const int tEnd = (chunk + 1) * NT_PER;
    for (int t = chunk * NT_PER + wave; t < tEnd; t += WAVES) {
        const int row = t * 16 + mrow;
        const float* p = table + row * LAT;
        __builtin_prefetch(p + WAVES * 16 * LAT, 0, 3);
        v16h Bv;
#pragma unroll
        for (int i = 0; i < 16; ++i) Bv[i] = (_Float16)0.f;
        if (lo) {
            const float4 f0 = *(const float4*)(p + 0);
            const float4 f1 = *(const float4*)(p + 4);
            const float4 f2 = *(const float4*)(p + 8);
            const float4 f3 = *(const float4*)(p + 12);
            Bv[0]=(_Float16)f0.x; Bv[1]=(_Float16)f0.y; Bv[2]=(_Float16)f0.z; Bv[3]=(_Float16)f0.w;
            Bv[4]=(_Float16)f1.x; Bv[5]=(_Float16)f1.y; Bv[6]=(_Float16)f1.z; Bv[7]=(_Float16)f1.w;
            Bv[8]=(_Float16)f2.x; Bv[9]=(_Float16)f2.y; Bv[10]=(_Float16)f2.z; Bv[11]=(_Float16)f2.w;
            Bv[12]=(_Float16)f3.x; Bv[13]=(_Float16)f3.y; Bv[14]=(_Float16)f3.z; Bv[15]=(_Float16)f3.w;
        } else {
            const float4 f4 = *(const float4*)(p + 16);
            Bv[0]=(_Float16)f4.x; Bv[1]=(_Float16)f4.y; Bv[2]=(_Float16)f4.z; Bv[3]=(_Float16)f4.w;
        }
        v8f acc = {};
        acc = __builtin_amdgcn_wmma_f32_16x16x32_f16(false, Av, false, Bv, (short)0, acc, false, false);
        const float tnv = tn[row];
#pragma unroll
        for (int v = 0; v < 8; ++v) {
            float s = fmaf(-2.f, acc[v], tnv);
            if (s < mn[v]) { mn[v] = s; ix[v] = row; }
        }
    }
#pragma unroll
    for (int v = 0; v < 8; ++v) { sMin[wave][v][lane] = mn[v]; sIdx[wave][v][lane] = ix[v]; }
    __syncthreads();
    reduce_and_emit(sMin, sIdx, tid, mtile, chunk, pmin, pidx);
}

// ---------------- Phase 2: finalize keys + label swap ----------------
__global__ void knn_phase2(
    const float* __restrict__ mius, const float* __restrict__ table,
    const float* __restrict__ softlabels, const float* __restrict__ new_softlabels,
    const int* __restrict__ flip_table, const int* __restrict__ flip_offset,
    const float* __restrict__ pmin, const int* __restrict__ pidx,
    float* __restrict__ out)
{
    const int b = blockIdx.x * blockDim.x + threadIdx.x;
    if (b >= BATCH) return;

    float best = FLT_BIG; int nn = 0x7fffffff;
    for (int c = 0; c < NSPLIT; ++c) {
        float v = pmin[b * NSPLIT + c];
        int  id = pidx[b * NSPLIT + c];
        if (v < best || (v == best && id < nn)) { best = v; nn = id; }
    }

    // Exact f32 distance to the winner (hit queries are bitwise copies -> exactly 0).
    float s = 0.f;
#pragma unroll
    for (int k = 0; k < LAT; ++k) {
        float d = mius[b * LAT + k] - table[nn * LAT + k];
        s = fmaf(d, d, s);
    }
    const int key = (s <= 1e-12f) ? nn : -1;    // mind <= 1e-6  <=>  d2 <= 1e-12

    const float* sl = softlabels + b * NCLS;
    float mx = sl[0]; int t = 0;
    for (int c = 1; c < NCLS; ++c)
        if (sl[c] > mx) { mx = sl[c]; t = c; }

    const int off = (key != -1 && flip_table[b] == 1) ? flip_offset[b] : 0;
    const int f = (t + off) % NCLS;
    const bool member = (key != -1) && (flip_table[key] == 1);

    const float* ns = new_softlabels + b * NCLS;
    float* o = out + b * NCLS;
    for (int c = 0; c < NCLS; ++c) {
        int idx = c;
        if (member) { if (c == t) idx = f; else if (c == f) idx = t; }
        o[c] = ns[idx];
    }
}

extern "C" void kernel_launch(void* const* d_in, const int* in_sizes, int n_in,
                              void* d_out, int out_size, void* d_ws, size_t ws_size,
                              hipStream_t stream) {
    const float* mius           = (const float*)d_in[0];
    /* d_in[1] = logvars, unused by the reference */
    const float* softlabels     = (const float*)d_in[2];
    const float* new_softlabels = (const float*)d_in[3];
    const float* table_miu      = (const float*)d_in[4];
    const int*   flip_table     = (const int*)d_in[5];
    const int*   flip_offset    = (const int*)d_in[6];
    float* out = (float*)d_out;

    // Workspace layout: pmin | pidx | (packed f16 table  OR  tn array)
    float* pmin = (float*)d_ws;
    int*   pidx = (int*)(pmin + BATCH * NSPLIT);
    void*  rest = (void*)(pidx + BATCH * NSPLIT);
    const size_t head  = (size_t)BATCH * NSPLIT * 8;
    const size_t needP = head + (size_t)N_TAB * PKW * sizeof(_Float16); // ~13 MB

    dim3 g1(BATCH / 16, NSPLIT);
    if (ws_size >= needP) {
        _Float16* pk = (_Float16*)rest;
        pack_kernel<<<(N_TAB + 255) / 256, 256, 0, stream>>>(table_miu, pk);
        knn_phase1_packed<<<g1, 256, 0, stream>>>(mius, pk, pmin, pidx);
    } else {
        float* tnArr = (float*)rest;
        tn_kernel<<<(N_TAB + 255) / 256, 256, 0, stream>>>(table_miu, tnArr);
        knn_phase1_fallback<<<g1, 256, 0, stream>>>(mius, table_miu, tnArr, pmin, pidx);
    }

    knn_phase2<<<(BATCH + 127) / 128, 128, 0, stream>>>(
        mius, table_miu, softlabels, new_softlabels,
        flip_table, flip_offset, pmin, pidx, out);
}
